// Mukara_45698452030097
// MI455X (gfx1250) — matrix operand order repaired and checked by reading, not compiled, not run on hardware
//
#include <hip/hip_runtime.h>
#include <hip/hip_bf16.h>

// ---------------------------------------------------------------------------
// CDNA5 (gfx1250) line-graph GNN.
// Precision: bf16 WMMA (V_WMMA_F32_16X16X32_BF16) with fp32 accumulate for all
// GEMMs; fp32 everywhere else (gather/scatter/LayerNorm/head reduction).
// ---------------------------------------------------------------------------

typedef __attribute__((ext_vector_type(16))) __bf16 v16bf;
typedef __attribute__((ext_vector_type(8)))  float  v8f;

#define D_DIM 128

// ---------------------------------------------------------------------------
// Pack a fp32 row-major weight [Ksrc,128] into bf16 B-fragment-major layout:
//   dst[(((nt*(Kp/32)+ks)*32 + lane)*16 + j]  where
//   n = nt*16 + (lane&15),  k = ks*32 + 16*(jv>>2) + 2*(jv&3) + 8*(lane>>4) + (j&1), jv=j>>1
// (same k order as the A fragment a_k: k = 8*half + j for j<8, 16 + 8*half + (j-8) for j>=8)
// Rows k >= Ksrc are zero-padded (used to pad F_IN=16 up to K=32).
// ---------------------------------------------------------------------------
__global__ __launch_bounds__(256) void k_pack(const float* __restrict__ src,
                                              __bf16* __restrict__ dst,
                                              int Kp, int Ksrc) {
    int i = blockIdx.x * 256 + threadIdx.x;
    int total = 128 * Kp;                 // 8 ntiles * (Kp/32) ksteps * 32 * 16
    if (i >= total) return;
    int j    = i & 15;
    int lane = (i >> 4) & 31;
    int rest = i >> 9;
    int nk   = Kp >> 5;
    int ks   = rest % nk;
    int nt   = rest / nk;
    int jv   = j >> 1;
    int k    = ks * 32 + 16 * (jv >> 2) + 2 * (jv & 3) + 8 * (lane >> 4) + (j & 1);
    int n    = nt * 16 + (lane & 15);
    float v  = (k < Ksrc) ? src[k * D_DIM + n] : 0.0f;
    dst[i] = (__bf16)v;
}

__global__ __launch_bounds__(256) void k_zero(float* __restrict__ p, int n) {
    int i = blockIdx.x * 256 + threadIdx.x;
    if (i < n) p[i] = 0.0f;
}

__global__ __launch_bounds__(256) void k_count(const int* __restrict__ dst,
                                               float* __restrict__ counts, int M) {
    int i = blockIdx.x * 256 + threadIdx.x;
    if (i < M) atomicAdd(&counts[dst[i]], 1.0f);
}

__global__ __launch_bounds__(256) void k_invcnt(const float* __restrict__ counts,
                                                float* __restrict__ inv, int E) {
    int i = blockIdx.x * 256 + threadIdx.x;
    if (i < E) {
        float c = counts[i];
        inv[i] = (c > 0.0f) ? (1.0f / c) : 0.0f;
    }
}

// One lane per (edge, feature): lg_src/lg_dst loads are wave-uniform, h reads
// and agg atomics are fully coalesced; agg stays L2-resident (51 MB < 192 MB).
__global__ __launch_bounds__(256) void k_scatter(const float* __restrict__ H,
                                                 const int* __restrict__ src,
                                                 const int* __restrict__ dst,
                                                 float* __restrict__ AGG, int total) {
    int i = blockIdx.x * 256 + threadIdx.x;
    if (i >= total) return;
    int e = i >> 7;
    int d = i & 127;
    int s = src[e];
    int t = dst[e];
    atomicAdd(&AGG[(size_t)t * D_DIM + d], H[(size_t)s * D_DIM + d]);
}

// ---------------------------------------------------------------------------
// A-fragment index helper (ISA 7.12.2, 16-bit A 16x32):
//   element j (j=0..15): jv=j>>1, k = 16*(jv>>2) + 2*(jv&3) + 8*half + (j&1)
// ---------------------------------------------------------------------------
__device__ __forceinline__ int a_k(int j, int half) {
    int jv = j >> 1;
    return 16 * (jv >> 2) + 2 * (jv & 3) + 8 * half + (j & 1);
}

// ---------------------------------------------------------------------------
// edge_proj: H = relu(X[16x16 pad->32] @ W1[32x128] + b1) @ W2[128x128] + b2
// One block = 16 rows; 8 waves = 8 column tiles of 16.
// ---------------------------------------------------------------------------
__global__ __launch_bounds__(256) void k_proj(const float* __restrict__ X,
                                              const __bf16* __restrict__ W1p,
                                              const float* __restrict__ b1,
                                              const __bf16* __restrict__ W2p,
                                              const float* __restrict__ b2,
                                              float* __restrict__ H) {
    __shared__ __bf16 t_lds[16 * D_DIM];
    const int tid  = threadIdx.x;
    const int lane = tid & 31, wave = tid >> 5;
    const int l16  = lane & 15, half = lane >> 4;
    const int row0 = blockIdx.x * 16;

    // GEMM1 (K padded 16->32, single k-step)
    v16bf a;
    const int m = row0 + l16;
#pragma unroll
    for (int j = 0; j < 16; ++j) {
        int k = a_k(j, half);
        a[j] = (k < 16) ? (__bf16)X[m * 16 + k] : (__bf16)0.0f;
    }
    v16bf b = *(const v16bf*)(W1p + ((size_t)wave * 32 + lane) * 16);
    v8f c = {};
    c = __builtin_amdgcn_wmma_f32_16x16x32_bf16(false, a, false, b, (short)0, c,
                                                false, false);
    {
        float bias = b1[wave * 16 + l16];
#pragma unroll
        for (int j = 0; j < 8; ++j) {
            float v = c[j] + bias;
            t_lds[(j + 8 * half) * D_DIM + wave * 16 + l16] =
                (__bf16)(v > 0.0f ? v : 0.0f);
        }
    }
    __syncthreads();

    // GEMM2 (K=128, 4 k-steps)
    v8f acc = {};
#pragma unroll
    for (int ks = 0; ks < 4; ++ks) {
        v16bf a2;
#pragma unroll
        for (int j = 0; j < 16; ++j)
            a2[j] = t_lds[l16 * D_DIM + ks * 32 + a_k(j, half)];
        v16bf bf = *(const v16bf*)(W2p + (((size_t)wave * 4 + ks) * 32 + lane) * 16);
        acc = __builtin_amdgcn_wmma_f32_16x16x32_bf16(false, a2, false, bf, (short)0,
                                                      acc, false, false);
    }
    {
        float bias = b2[wave * 16 + l16];
#pragma unroll
        for (int j = 0; j < 8; ++j)
            H[(size_t)(row0 + j + 8 * half) * D_DIM + wave * 16 + l16] = acc[j] + bias;
    }
}

// ---------------------------------------------------------------------------
// One GNN layer, fused: z=[h | agg*inv] -> relu(z@W1+b1)@W2+b2 -> +h -> LN.
// Block owns 16 rows; z (bf16) / t (bf16) / y (f32) staged in LDS.
// ---------------------------------------------------------------------------
__global__ __launch_bounds__(256) void k_update(float* __restrict__ H,
                                                const float* __restrict__ AGG,
                                                const float* __restrict__ INV,
                                                const __bf16* __restrict__ W1p,
                                                const float* __restrict__ b1,
                                                const __bf16* __restrict__ W2p,
                                                const float* __restrict__ b2,
                                                const float* __restrict__ lng,
                                                const float* __restrict__ lnb) {
    __shared__ __bf16 z_lds[16 * 256];
    __shared__ __bf16 t_lds[16 * D_DIM];
    __shared__ float  y_lds[16 * D_DIM];
    __shared__ float  red[16 * 16];

    const int tid  = threadIdx.x;
    const int lane = tid & 31, wave = tid >> 5;
    const int l16  = lane & 15, half = lane >> 4;
    const int row0 = blockIdx.x * 16;

    // Stage z = [h | agg * inv_cnt], fp32 -> bf16, fully coalesced.
    for (int idx = tid; idx < 16 * 256; idx += 256) {
        int r = idx >> 8, cc = idx & 255;
        int gr = row0 + r;
        float v = (cc < D_DIM) ? H[(size_t)gr * D_DIM + cc]
                               : AGG[(size_t)gr * D_DIM + (cc - D_DIM)] * INV[gr];
        z_lds[idx] = (__bf16)v;
    }
    __syncthreads();

    // GEMM1: t = relu(z[16x256] @ W1[256x128] + b1), 8 k-steps
    v8f c = {};
#pragma unroll
    for (int ks = 0; ks < 8; ++ks) {
        v16bf a;
#pragma unroll
        for (int j = 0; j < 16; ++j)
            a[j] = z_lds[l16 * 256 + ks * 32 + a_k(j, half)];
        v16bf b = *(const v16bf*)(W1p + (((size_t)wave * 8 + ks) * 32 + lane) * 16);
        c = __builtin_amdgcn_wmma_f32_16x16x32_bf16(false, a, false, b, (short)0, c,
                                                    false, false);
    }
    {
        float bias = b1[wave * 16 + l16];
#pragma unroll
        for (int j = 0; j < 8; ++j) {
            float v = c[j] + bias;
            t_lds[(j + 8 * half) * D_DIM + wave * 16 + l16] =
                (__bf16)(v > 0.0f ? v : 0.0f);
        }
    }
    __syncthreads();

    // GEMM2: h_new = t[16x128] @ W2[128x128] + b2, then y = h + h_new
    v8f acc = {};
#pragma unroll
    for (int ks = 0; ks < 4; ++ks) {
        v16bf a2;
#pragma unroll
        for (int j = 0; j < 16; ++j)
            a2[j] = t_lds[l16 * D_DIM + ks * 32 + a_k(j, half)];
        v16bf bf = *(const v16bf*)(W2p + (((size_t)wave * 4 + ks) * 32 + lane) * 16);
        acc = __builtin_amdgcn_wmma_f32_16x16x32_bf16(false, a2, false, bf, (short)0,
                                                      acc, false, false);
    }
    {
        float bias = b2[wave * 16 + l16];
#pragma unroll
        for (int j = 0; j < 8; ++j) {
            int rr = j + 8 * half;
            int cc = wave * 16 + l16;
            y_lds[rr * D_DIM + cc] =
                H[(size_t)(row0 + rr) * D_DIM + cc] + acc[j] + bias;
        }
    }
    __syncthreads();

    // LayerNorm over D=128 per row: 16 threads per row, 8 cols each.
    const int r = tid >> 4, s = tid & 15;
    float part = 0.0f;
#pragma unroll
    for (int q = 0; q < 8; ++q) part += y_lds[r * D_DIM + s * 8 + q];
    red[r * 16 + s] = part;
    __syncthreads();
    float mean = 0.0f;
#pragma unroll
    for (int q = 0; q < 16; ++q) mean += red[r * 16 + q];
    mean *= (1.0f / (float)D_DIM);
    __syncthreads();
    float part2 = 0.0f;
#pragma unroll
    for (int q = 0; q < 8; ++q) {
        float d = y_lds[r * D_DIM + s * 8 + q] - mean;
        part2 += d * d;
    }
    red[r * 16 + s] = part2;
    __syncthreads();
    float var = 0.0f;
#pragma unroll
    for (int q = 0; q < 16; ++q) var += red[r * 16 + q];
    var *= (1.0f / (float)D_DIM);
    float rstd = rsqrtf(var + 1e-6f);
#pragma unroll
    for (int q = 0; q < 8; ++q) {
        int col = s * 8 + q;
        float v = (y_lds[r * D_DIM + col] - mean) * rstd * lng[col] + lnb[col];
        H[(size_t)(row0 + r) * D_DIM + col] = v;
    }
}

// ---------------------------------------------------------------------------
// Output head: out = relu(H @ out_w1 + b1) @ out_w2 + b2  -> [E,1]
// ---------------------------------------------------------------------------
__global__ __launch_bounds__(256) void k_out(const float* __restrict__ H,
                                             const __bf16* __restrict__ W1p,
                                             const float* __restrict__ b1,
                                             const float* __restrict__ w2,
                                             const float* __restrict__ b2,
                                             float* __restrict__ OUT) {
    __shared__ float t_lds[16 * D_DIM];
    __shared__ float red[16 * 16];
    const int tid  = threadIdx.x;
    const int lane = tid & 31, wave = tid >> 5;
    const int l16  = lane & 15, half = lane >> 4;
    const int row0 = blockIdx.x * 16;

    v8f acc = {};
#pragma unroll
    for (int ks = 0; ks < 4; ++ks) {
        v16bf a;
#pragma unroll
        for (int j = 0; j < 16; ++j)
            a[j] = (__bf16)H[(size_t)(row0 + l16) * D_DIM + ks * 32 + a_k(j, half)];
        v16bf b = *(const v16bf*)(W1p + (((size_t)wave * 4 + ks) * 32 + lane) * 16);
        acc = __builtin_amdgcn_wmma_f32_16x16x32_bf16(false, a, false, b, (short)0,
                                                      acc, false, false);
    }
    {
        float bias = b1[wave * 16 + l16];
#pragma unroll
        for (int j = 0; j < 8; ++j) {
            float v = acc[j] + bias;
            t_lds[(j + 8 * half) * D_DIM + wave * 16 + l16] = v > 0.0f ? v : 0.0f;
        }
    }
    __syncthreads();
    const int r = tid >> 4, s = tid & 15;
    float part = 0.0f;
#pragma unroll
    for (int q = 0; q < 8; ++q) {
        int col = s * 8 + q;
        part += t_lds[r * D_DIM + col] * w2[col];
    }
    red[r * 16 + s] = part;
    __syncthreads();
    if (s == 0) {
        float o = b2[0];
#pragma unroll
        for (int q = 0; q < 16; ++q) o += red[r * 16 + q];
        OUT[row0 + r] = o;
    }
}

// ---------------------------------------------------------------------------
extern "C" void kernel_launch(void* const* d_in, const int* in_sizes, int n_in,
                              void* d_out, int out_size, void* d_ws, size_t ws_size,
                              hipStream_t stream) {
    const float* edge_features = (const float*)d_in[0];
    const int*   lg_src        = (const int*)d_in[1];
    const int*   lg_dst        = (const int*)d_in[2];
    const float* proj_w1       = (const float*)d_in[3];
    const float* proj_b1       = (const float*)d_in[4];
    const float* proj_w2       = (const float*)d_in[5];
    const float* proj_b2       = (const float*)d_in[6];
    const float* up_w1         = (const float*)d_in[7];
    const float* up_b1         = (const float*)d_in[8];
    const float* up_w2         = (const float*)d_in[9];
    const float* up_b2         = (const float*)d_in[10];
    const float* ln_g          = (const float*)d_in[11];
    const float* ln_b          = (const float*)d_in[12];
    const float* out_w1        = (const float*)d_in[13];
    const float* out_b1        = (const float*)d_in[14];
    const float* out_w2        = (const float*)d_in[15];
    const float* out_b2        = (const float*)d_in[16];

    const int E = in_sizes[0] / 16;        // 100000 (multiple of 16)
    const int M = in_sizes[1];             // 1600000
    const int L = in_sizes[8] / D_DIM;     // 3

    // Workspace carve-out (256B aligned): ~103.6 MB total.
    char* p = (char*)d_ws;
    auto carve = [&](size_t bytes) -> char* {
        char* r = p;
        p += (bytes + 255) & ~(size_t)255;
        return r;
    };
    float*  h      = (float*)carve((size_t)E * D_DIM * sizeof(float));
    float*  agg    = (float*)carve((size_t)E * D_DIM * sizeof(float));
    float*  counts = (float*)carve((size_t)E * sizeof(float));
    float*  inv    = (float*)carve((size_t)E * sizeof(float));
    __bf16* w1p    = (__bf16*)carve((size_t)128 * 32  * 2);
    __bf16* w2p    = (__bf16*)carve((size_t)128 * 128 * 2);
    __bf16* u1p    = (__bf16*)carve((size_t)L * 128 * 256 * 2);
    __bf16* u2p    = (__bf16*)carve((size_t)L * 128 * 128 * 2);
    __bf16* o1p    = (__bf16*)carve((size_t)128 * 128 * 2);

    // Weight packing (bf16, B-fragment layout)
    k_pack<<<(128 * 32 + 255) / 256, 256, 0, stream>>>(proj_w1, w1p, 32, 16);
    k_pack<<<(128 * 128 + 255) / 256, 256, 0, stream>>>(proj_w2, w2p, 128, 128);
    for (int i = 0; i < L; ++i) {
        k_pack<<<(128 * 256 + 255) / 256, 256, 0, stream>>>(
            up_w1 + (size_t)i * 256 * D_DIM, u1p + (size_t)i * 128 * 256, 256, 256);
        k_pack<<<(128 * 128 + 255) / 256, 256, 0, stream>>>(
            up_w2 + (size_t)i * D_DIM * D_DIM, u2p + (size_t)i * 128 * 128, 128, 128);
    }
    k_pack<<<(128 * 128 + 255) / 256, 256, 0, stream>>>(out_w1, o1p, 128, 128);

    // In-degree counts -> inv_cnt
    k_zero<<<(E + 255) / 256, 256, 0, stream>>>(counts, E);
    k_count<<<(M + 255) / 256, 256, 0, stream>>>(lg_dst, counts, M);
    k_invcnt<<<(E + 255) / 256, 256, 0, stream>>>(counts, inv, E);

    // edge_proj
    k_proj<<<E / 16, 256, 0, stream>>>(edge_features, w1p, proj_b1, w2p, proj_b2, h);

    // GNN layers
    const int nAgg = E * D_DIM;
    const int nSc  = M * D_DIM;            // 204.8M lanes, fits in int
    for (int i = 0; i < L; ++i) {
        k_zero<<<(nAgg + 255) / 256, 256, 0, stream>>>(agg, nAgg);
        k_scatter<<<(nSc + 255) / 256, 256, 0, stream>>>(h, lg_src, lg_dst, agg, nSc);
        k_update<<<E / 16, 256, 0, stream>>>(
            h, agg, inv,
            u1p + (size_t)i * 128 * 256, up_b1 + (size_t)i * D_DIM,
            u2p + (size_t)i * 128 * 128, up_b2 + (size_t)i * D_DIM,
            ln_g + (size_t)i * D_DIM, ln_b + (size_t)i * D_DIM);
    }

    // Output head
    k_out<<<E / 16, 256, 0, stream>>>(h, o1p, out_b1, out_w2, out_b2, (float*)d_out);
}
